// PN_head_22110491639917
// MI455X (gfx1250) — matrix operand order repaired and checked by reading, not compiled, not run on hardware
//
#include <hip/hip_runtime.h>
#include <hip/hip_bf16.h>
#include <stdint.h>

#define NS   16384
#define NQ   8192
#define DIM  1024
#define NC   64
#define SELF_EPS 5e-7f

typedef __attribute__((ext_vector_type(16))) __bf16 bf16x16;
typedef __attribute__((ext_vector_type(8)))  float  fx8;
typedef __attribute__((ext_vector_type(4)))  unsigned u32x4;
typedef __attribute__((ext_vector_type(8)))  unsigned u32x8;

// ---- WMMA fragment gathers (CDNA5 16-bit A/B layouts, CDNA5 ISA §7.12.2) ----
// Row pointers carry the per-lane `half` offset so the k-loop emits pure
// ds_load_b128 [vRow] offset:imm with no per-iteration address VALU.
__device__ __forceinline__ bf16x16 fragA_at(const __bf16* p, int k0) {
    union { bf16x16 v; uint4 q[2]; } f;
    f.q[0] = *(const uint4*)(p + k0);        // K=[8h,8h+8)
    f.q[1] = *(const uint4*)(p + k0 + 16);   // K=[16+8h,16+8h+8)
    return f.v;
}
__device__ __forceinline__ bf16x16 fragB_at(const __bf16* p, int k0) {
    union { bf16x16 v; uint4 q2[2]; } f;
    f.q2[0] = *(const uint4*)(p + k0);       // K=[8h,8h+8)
    f.q2[1] = *(const uint4*)(p + k0 + 16);  // K=[16+8h,16+8h+8)
    return f.v;
}

// ---- Tensor Data Mover: DMA one 32-proto x 1024-bf16 chunk (64KB) to LDS ----
// D# group0: [1:0]=count=1 | [63:32]=lds_addr | [120:64]=global_addr | [127:126]=type=2
// D# group1: data_size=4B; tensor/tile = 512 dwords x 32 rows, stride 512.
__device__ __forceinline__ void tdm_load_chunk(const __bf16* gsrc, __bf16* ldst) {
    unsigned long long ga = (unsigned long long)(uintptr_t)gsrc;
    unsigned lds = (unsigned)(uintptr_t)ldst;   // generic addr[31:0] == LDS byte addr
    u32x4 g0;
    g0[0] = 1u;                                  // count=1 (valid user descriptor)
    g0[1] = lds;                                 // lds_addr
    g0[2] = (unsigned)ga;                        // global_addr[31:0]
    g0[3] = ((unsigned)(ga >> 32) & 0x01ffffffu) | (2u << 30); // ga[56:32] | type=2
    u32x8 g1;
    g1[0] = 0x00020000u;                         // wg_mask=0, data_size=2 (4B), no pad
    g1[1] = (512u & 0xffffu) << 16;              // tensor_dim0[15:0] = 512 dwords
    g1[2] = (512u >> 16) | (32u << 16);          // tensor_dim0 hi | tensor_dim1 lo = 32
    g1[3] = (32u >> 16) | (512u << 16);          // tensor_dim1 hi | tile_dim0 = 512
    g1[4] = 32u;                                 // tile_dim1 = 32, tile_dim2 = 0
    g1[5] = 512u;                                // tensor_dim0_stride lo = 512
    g1[6] = 0u;                                  // stride0 hi | stride1 lo (unused, 2D)
    g1[7] = 0u;
    asm volatile("tensor_load_to_lds %0, %1" :: "s"(g0), "s"(g1) : "memory");
}

// ---- K0: zero segment-sum workspace ----
__global__ void zero_ws(float* p, int n) {
    int i = blockIdx.x * blockDim.x + threadIdx.x;
    if (i < n) p[i] = 0.0f;
}

// ---- K1: L2-normalize rows -> bf16; optionally fuse per-class segment sums ----
__global__ void __launch_bounds__(256) normalize_rows(
    const float* __restrict__ x, __bf16* __restrict__ y, int rows,
    const int* __restrict__ labels, float* __restrict__ psum, float* __restrict__ pcnt)
{
    const int wv = threadIdx.x >> 5, lane = threadIdx.x & 31;
    const int row = blockIdx.x * 8 + wv;
    if (row >= rows) return;
    const float* xr = x + (size_t)row * DIM;
    float s = 0.0f;
    for (int i = lane; i < DIM; i += 32) { float v = xr[i]; s += v * v; }
    #pragma unroll
    for (int m = 16; m; m >>= 1) s += __shfl_xor(s, m, 32);
    const float inv = 1.0f / fmaxf(sqrtf(s), 1e-12f);
    __bf16* yr = y + (size_t)row * DIM;
    if (labels) {
        const int lab = labels[row];
        float* ps = psum + (size_t)lab * DIM;
        for (int i = lane; i < DIM; i += 32) {
            float v = xr[i] * inv;
            yr[i] = (__bf16)v;
            atomicAdd(ps + i, v);
        }
        if (lane == 0) atomicAdd(pcnt + lab, 1.0f);
    } else {
        for (int i = lane; i < DIM; i += 32) yr[i] = (__bf16)(xr[i] * inv);
    }
}

// ---- K3: prototypes = l2norm(segment_mean) -> bf16 ----
__global__ void __launch_bounds__(256) build_protos(
    const float* __restrict__ psum, const float* __restrict__ pcnt, __bf16* __restrict__ proto)
{
    __shared__ float red[256];
    const int c = blockIdx.x, tid = threadIdx.x;
    const float cnt = fmaxf(pcnt[c], 1.0f);
    const float* pr = psum + (size_t)c * DIM;
    float s = 0.0f;
    for (int i = tid; i < DIM; i += 256) { float m = pr[i] / cnt; s += m * m; }
    red[tid] = s; __syncthreads();
    for (int off = 128; off; off >>= 1) {
        if (tid < off) red[tid] += red[tid + off];
        __syncthreads();
    }
    const float inv = 1.0f / fmaxf(sqrtf(red[0]), 1e-12f);
    for (int i = tid; i < DIM; i += 256)
        proto[(size_t)c * DIM + i] = (__bf16)((pr[i] / cnt) * inv);
}

// ---- K4: fused sim GEMM (bf16 WMMA) + per-query running top-6 ----
// 64 queries x all NS protos per block. A resident in LDS (128KB); B chunks
// (32 protos x full K, 64KB) double-buffered and streamed by the TDM while
// the 32-step pipelined WMMA k-loop computes the previous chunk.
__global__ void __launch_bounds__(256) knn_gemm_topk(
    const __bf16* __restrict__ qn, const __bf16* __restrict__ pn,
    float* __restrict__ tv_out, int* __restrict__ ti_out)
{
    extern __shared__ __align__(16) char smem[];
    __bf16* Bs0 = (__bf16*)smem;                                  // 32 x 1024
    __bf16* Bs1 = (__bf16*)(smem + (size_t)32 * DIM * 2);         // 32 x 1024
    __bf16* As  = (__bf16*)(smem + (size_t)64 * DIM * 2);         // 64 x 1024
    float*  Sm  = (float*)(smem + (size_t)128 * DIM * 2);         // 64 x 32

    const int tid = threadIdx.x;
    const int q0  = blockIdx.x * 64;
    const int lane = tid & 31, wv = tid >> 5;
    const int mt = (wv & 3) * 16;       // 4 M-tiles
    const int nt = (wv >> 2) * 16;      // 2 N-tiles
    const int half = lane >> 4, lrow = lane & 15;

    // prologue: TDM chunk 0 -> Bs0 while all threads stage the A block
    if (wv == 0) tdm_load_chunk(pn, Bs0);
    {
        const uint4* src = (const uint4*)(qn + (size_t)q0 * DIM);
        uint4* dst = (uint4*)As;
        for (int i = tid; i < 64 * DIM / 8; i += 256) dst[i] = src[i];
    }

    float tv[6]; int ti[6];
    #pragma unroll
    for (int j = 0; j < 6; ++j) { tv[j] = -1e30f; ti[j] = 0; }

    const __bf16* Arow = As + (size_t)(mt + lrow) * DIM + half * 8;

    for (int n0 = 0; n0 < NS; n0 += 32) {
        __bf16* Bcur = ((n0 >> 5) & 1) ? Bs1 : Bs0;
        if (wv == 0) __builtin_amdgcn_s_wait_tensorcnt(0);  // chunk n0 landed
        __syncthreads();   // publish Bcur; prior chunk's consumers are done
        if (wv == 0 && n0 + 32 < NS)                        // overlap next DMA
            tdm_load_chunk(pn + (size_t)(n0 + 32) * DIM,
                           (((n0 >> 5) + 1) & 1) ? Bs1 : Bs0);

        const __bf16* Brow = Bcur + (size_t)(nt + lrow) * DIM + half * 8;

        // software-pipelined k-loop: loads for k+32 in flight under WMMA at k
        fx8 acc = {0.f, 0.f, 0.f, 0.f, 0.f, 0.f, 0.f, 0.f};
        bf16x16 a = fragA_at(Arow, 0);
        bf16x16 b = fragB_at(Brow, 0);
        #pragma unroll 4
        for (int k0 = 32; k0 < DIM; k0 += 32) {
            bf16x16 an = fragA_at(Arow, k0);
            bf16x16 bn = fragB_at(Brow, k0);
            acc = __builtin_amdgcn_wmma_f32_16x16x32_bf16(
                false, a, false, b, (short)0, acc, false, false);
            a = an; b = bn;
        }
        acc = __builtin_amdgcn_wmma_f32_16x16x32_bf16(
            false, a, false, b, (short)0, acc, false, false);

        // spill 16x16 tile: VGPR r, lanes0-15 -> M=r, lanes16-31 -> M=r+8
        #pragma unroll
        for (int r = 0; r < 8; ++r)
            Sm[(mt + r + half * 8) * 32 + nt + lrow] = acc[r];
        __syncthreads();

        if (tid < 64) {                  // thread t owns query row t: update top-6
            #pragma unroll 4
            for (int j = 0; j < 32; ++j) {
                const float v = Sm[tid * 32 + j];
                if (v > tv[5]) {
                    tv[5] = v; ti[5] = n0 + j;
                    #pragma unroll
                    for (int s = 5; s > 0; --s)
                        if (tv[s] > tv[s - 1]) {
                            float t = tv[s]; tv[s] = tv[s - 1]; tv[s - 1] = t;
                            int   u = ti[s]; ti[s] = ti[s - 1]; ti[s - 1] = u;
                        }
                }
            }
        }
    }
    if (tid < 64) {
        const int q = q0 + tid;
        #pragma unroll
        for (int j = 0; j < 6; ++j) { tv_out[q * 6 + j] = tv[j]; ti_out[q * 6 + j] = ti[j]; }
    }
}

// ---- K5: scores = qn @ protos.T (8192x64, K=1024) via WMMA ----
__global__ void __launch_bounds__(256) scores_gemm(
    const __bf16* __restrict__ qn, const __bf16* __restrict__ proto,
    const float* __restrict__ scale, float* __restrict__ out_cls, float* __restrict__ out_sc)
{
    __shared__ __align__(16) __bf16 As[32 * 32];
    __shared__ __align__(16) __bf16 Bs[64 * 32];
    const int tid = threadIdx.x;
    const int q0 = blockIdx.x * 32;
    const int lane = tid & 31, wv = tid >> 5;
    const int mt = (wv & 1) * 16, nt = (wv >> 1) * 16;
    const int half = lane >> 4, lrow = lane & 15;
    const float s = scale[0];

    fx8 acc = {0.f, 0.f, 0.f, 0.f, 0.f, 0.f, 0.f, 0.f};
    for (int k0 = 0; k0 < DIM; k0 += 32) {
        __syncthreads();
        for (int i = tid; i < 384; i += 256) {
            if (i < 128) {
                int r = i >> 2, c = (i & 3) * 8;
                *(uint4*)&As[r * 32 + c] = *(const uint4*)(qn + (size_t)(q0 + r) * DIM + k0 + c);
            } else {
                int j = i - 128; int r = j >> 2, c = (j & 3) * 8;
                *(uint4*)&Bs[r * 32 + c] = *(const uint4*)(proto + (size_t)r * DIM + k0 + c);
            }
        }
        if (k0 + 32 < DIM)   // global_prefetch_b8 on next K slab
            __builtin_prefetch(qn + (size_t)(q0 + (tid >> 3)) * DIM + k0 + 32, 0, 1);
        __syncthreads();
        bf16x16 a = fragA_at(As + (size_t)(mt + lrow) * 32 + half * 8, 0);
        bf16x16 b = fragB_at(Bs + (size_t)(nt + lrow) * 32 + half * 8, 0);
        acc = __builtin_amdgcn_wmma_f32_16x16x32_bf16(
            false, a, false, b, (short)0, acc, false, false);
    }
    #pragma unroll
    for (int r = 0; r < 8; ++r) {
        const int m = q0 + mt + r + half * 8;
        const int n = nt + lrow;
        const float v = acc[r];
        out_sc[(size_t)m * NC + n]  = v;
        out_cls[(size_t)m * NC + n] = s * v;
    }
}

// ---- K6: self-exclusion window, kNN vote (argmax = smallest class on tie) ----
__global__ void __launch_bounds__(256) finalize_topk(
    const float* __restrict__ tv, const int* __restrict__ ti,
    const int* __restrict__ labels,
    float* __restrict__ dist2, int* __restrict__ idx2, int* __restrict__ preds)
{
    const int q = blockIdx.x * blockDim.x + threadIdx.x;
    if (q >= NQ) return;
    const float v0 = tv[q * 6 + 0];
    const int off = (fabsf(1.0f - v0) < SELF_EPS) ? 1 : 0;
    int lab[5];
    #pragma unroll
    for (int j = 0; j < 5; ++j) {
        const int id = ti[q * 6 + j + off];
        dist2[q * 5 + j] = 1.0f - tv[q * 6 + j + off];
        idx2[q * 5 + j]  = id;
        lab[j] = labels[id];
    }
    int best_lab = 0x7fffffff, best_cnt = 0;
    #pragma unroll
    for (int j = 0; j < 5; ++j) {
        int c = 0;
        #pragma unroll
        for (int l = 0; l < 5; ++l) c += (lab[l] == lab[j]) ? 1 : 0;
        if (c > best_cnt || (c == best_cnt && lab[j] < best_lab)) { best_cnt = c; best_lab = lab[j]; }
    }
    preds[q] = best_lab;
}

extern "C" void kernel_launch(void* const* d_in, const int* in_sizes, int n_in,
                              void* d_out, int out_size, void* d_ws, size_t ws_size,
                              hipStream_t stream)
{
    (void)in_sizes; (void)n_in; (void)out_size; (void)ws_size;
    const float* p_img  = (const float*)d_in[0];
    const int*   p_lab  = (const int*)d_in[1];
    const float* q_img  = (const float*)d_in[2];
    const float* scale  = (const float*)d_in[5];   // d_in[3],[4] (support_*) unused by reference

    char* ws = (char*)d_ws;
    __bf16* qn    = (__bf16*)ws;  ws += (size_t)NQ * DIM * 2;
    __bf16* pn    = (__bf16*)ws;  ws += (size_t)NS * DIM * 2;
    float*  psum  = (float*)ws;   ws += (size_t)NC * DIM * 4;
    float*  pcnt  = (float*)ws;   ws += 256;
    __bf16* proto = (__bf16*)ws;  ws += (size_t)NC * DIM * 2;
    float*  tvv   = (float*)ws;   ws += (size_t)NQ * 6 * 4;
    int*    tii   = (int*)ws;

    float* out     = (float*)d_out;
    float* out_cls = out;                       // 8192*64
    int*   out_idx = (int*)(out + 524288);      // 8192*5
    float* out_dst = out + 565248;              // 8192*5
    int*   out_prd = (int*)(out + 606208);      // 8192
    float* out_sco = out + 614400;              // 8192*64

    zero_ws<<<(NC * DIM + 64 + 255) / 256, 256, 0, stream>>>(psum, NC * DIM + 64);
    normalize_rows<<<NS / 8, 256, 0, stream>>>(p_img, pn, NS, p_lab, psum, pcnt);
    normalize_rows<<<NQ / 8, 256, 0, stream>>>(q_img, qn, NQ, nullptr, nullptr, nullptr);
    build_protos<<<NC, 256, 0, stream>>>(psum, pcnt, proto);

    // LDS: 2x64KB B (TDM double-buffer) + 128KB A + 8KB sim = 264KB (<320KB WGP)
    const size_t smem4 = (size_t)128 * DIM * 2 + (size_t)64 * 32 * 4;
    knn_gemm_topk<<<NQ / 64, 256, smem4, stream>>>(qn, pn, tvv, tii);

    scores_gemm<<<NQ / 32, 256, 0, stream>>>(qn, proto, scale, out_cls, out_sco);
    finalize_topk<<<NQ / 256, 256, 0, stream>>>(tvv, tii, p_lab, out_dst, out_idx, out_prd);
}
